// LightningIndexer_71545565217040
// MI455X (gfx1250) — matrix operand order repaired and checked by reading, not compiled, not run on hardware
//
#include <hip/hip_runtime.h>

// ---------------------------------------------------------------------------
// LightningIndexer on gfx1250 (MI455X): bf16 WMMA path, fp32 accumulate.
//   q = x @ Wq^T + bq ; k = x @ Wk^T + bk      (proj_kernel, WMMA bf16)
//   out[b,i,j] = sum_h w[h] * relu( q[b,i,h,:] . k[b,j,h,:] )  (scores_kernel)
// Scores kernel stages tiles with GLOBAL_LOAD_ASYNC_TO_LDS_B128 (ASYNCcnt).
// ---------------------------------------------------------------------------

typedef __bf16 bf16_t;
typedef __attribute__((ext_vector_type(16))) __bf16 v16bf;
typedef __attribute__((ext_vector_type(8)))  __bf16 v8bf;
typedef __attribute__((ext_vector_type(8)))  float  v8f;
typedef __attribute__((address_space(3))) const unsigned short* lds_cptr;

#define D_MODEL 1024
#define FAN_OUT 256     // HEADS * HDIM
#define HDIM    64
#define HEADS   4
#define SEQ     4096
#define MROWS   8192    // 2 * 4096 flattened (b, s)

// LDS byte address (wave-relative) of a __shared__ object
__device__ __forceinline__ unsigned lds_addr_of(const void* p) {
  return (unsigned)(unsigned long long)(lds_cptr)p;
}

// Async DMA: 16 bytes global -> LDS, tracked by ASYNCcnt (no VGPR data path).
__device__ __forceinline__ void async_copy16(unsigned lds_byte_addr, const void* gptr) {
  asm volatile("global_load_async_to_lds_b128 %0, %1, off"
               :: "v"(lds_byte_addr), "v"((unsigned long long)gptr)
               : "memory");
}
__device__ __forceinline__ void wait_async0() {
  asm volatile("s_wait_asynccnt 0x0" ::: "memory");
}

// A-fragment (16x32 MxK bf16), ISA 7.12.2 layout:
//   lanes 0-15 : M=lane,    halves 0..7 = K[0..7],  halves 8..15 = K[16..23]
//   lanes 16-31: M=lane-16, halves 0..7 = K[8..15], halves 8..15 = K[24..31]
__device__ __forceinline__ v16bf ldA(const unsigned short* base, int stride, int lane) {
  int m    = lane & 15;
  int koff = (lane >> 4) << 3;              // 0 or 8
  const unsigned short* p = base + m * stride + koff;
  v8bf lo = *(const v8bf*)(p);
  v8bf hi = *(const v8bf*)(p + 16);
  return __builtin_shufflevector(lo, hi, 0,1,2,3,4,5,6,7,8,9,10,11,12,13,14,15);
}

// B-fragment (32x16 KxN bf16): B[k][n] = Rows[n][k] (row-major source rows).
//   lanes 0-15 : N=lane,    halves 0..7 = K[0..7],  halves 8..15 = K[16..23]
//   lanes 16-31: N=lane-16, halves 0..7 = K[8..15], halves 8..15 = K[24..31]
__device__ __forceinline__ v16bf ldB(const unsigned short* base, int stride, int lane) {
  int n    = lane & 15;
  int koff = (lane >> 4) << 3;              // 0 or 8
  const unsigned short* p = base + n * stride + koff;
  v8bf lo = *(const v8bf*)(p);
  v8bf hi = *(const v8bf*)(p + 16);
  return __builtin_shufflevector(lo, hi, 0,1,2,3,4,5,6,7,8,9,10,11,12,13,14,15);
}

// ---------------------------------------------------------------------------
// Projection: OUT[m][n] = bias[n] + sum_k X[m][k] * W[n][k], stored as bf16.
// Macro-tile 64(M) x 64(N), K-chunks of 32.  8 waves x 2 tiles (shared A row).
// fp32 -> bf16 uses native casts (backend emits hardware cvt ops).
// ---------------------------------------------------------------------------
__global__ __launch_bounds__(256)
void proj_kernel(const float* __restrict__ x, const float* __restrict__ w,
                 const float* __restrict__ bias, unsigned short* __restrict__ outp) {
  __shared__ unsigned short lx[64 * 40];   // X tile, bf16, padded stride 40
  __shared__ unsigned short lw[64 * 40];   // W tile, bf16

  const int m0 = blockIdx.x * 64;
  const int n0 = blockIdx.y * 64;
  const int tid  = threadIdx.x;
  const int wave = tid >> 5;
  const int lane = tid & 31;
  const int ti   = wave >> 1;              // tile row   (0..3)
  const int tj0  = (wave & 1) * 2;         // tile cols  tj0, tj0+1

  const int nlan = lane & 15;
  const float b0 = bias[n0 + tj0 * 16 + nlan];
  const float b1 = bias[n0 + (tj0 + 1) * 16 + nlan];
  v8f acc0, acc1;
#pragma unroll
  for (int r = 0; r < 8; ++r) { acc0[r] = b0; acc1[r] = b1; }

  const int lrow = tid >> 2;               // 0..63
  const int lseg = (tid & 3) * 8;          // 0,8,16,24

  for (int k0 = 0; k0 < D_MODEL; k0 += 32) {
    __syncthreads();
    {
      const float* sx = x + (size_t)(m0 + lrow) * D_MODEL + k0 + lseg;
      const float* sw = w + (size_t)(n0 + lrow) * D_MODEL + k0 + lseg;
      float4 x0 = *(const float4*)(sx);
      float4 x1 = *(const float4*)(sx + 4);
      float4 w0 = *(const float4*)(sw);
      float4 w1 = *(const float4*)(sw + 4);
      v8bf px = { (bf16_t)x0.x, (bf16_t)x0.y, (bf16_t)x0.z, (bf16_t)x0.w,
                  (bf16_t)x1.x, (bf16_t)x1.y, (bf16_t)x1.z, (bf16_t)x1.w };
      v8bf pw = { (bf16_t)w0.x, (bf16_t)w0.y, (bf16_t)w0.z, (bf16_t)w0.w,
                  (bf16_t)w1.x, (bf16_t)w1.y, (bf16_t)w1.z, (bf16_t)w1.w };
      *(v8bf*)(lx + lrow * 40 + lseg) = px;
      *(v8bf*)(lw + lrow * 40 + lseg) = pw;
    }
    __syncthreads();

    v16bf a   = ldA(lx + ti  * 16 * 40, 40, lane);
    v16bf bb0 = ldB(lw + tj0 * 16 * 40, 40, lane);
    v16bf bb1 = ldB(lw + (tj0 + 1) * 16 * 40, 40, lane);
    acc0 = __builtin_amdgcn_wmma_f32_16x16x32_bf16(false, a, false, bb0, (short)0, acc0, false, false);
    acc1 = __builtin_amdgcn_wmma_f32_16x16x32_bf16(false, a, false, bb1, (short)0, acc1, false, false);
  }

  // C/D layout: M = r + 8*(lane>=16), N = lane&15
  bf16_t* ob = (bf16_t*)outp;
  const int mbase = m0 + ti * 16 + ((lane >> 4) << 3);
  const int nc0   = n0 + tj0 * 16 + nlan;
#pragma unroll
  for (int r = 0; r < 8; ++r) {
    ob[(size_t)(mbase + r) * FAN_OUT + nc0]      = (bf16_t)acc0[r];
    ob[(size_t)(mbase + r) * FAN_OUT + nc0 + 16] = (bf16_t)acc1[r];
  }
}

// ---------------------------------------------------------------------------
// Scores: out[b,i,j] = sum_h w[h] * relu( Q[b,i,h,:] . K[b,j,h,:] )
// Block = 64x64 output tile; per head, ASYNC-DMA Q/K head slices into LDS;
// each of 8 waves computes two 16x16 tiles via 2x wmma_f32_16x16x32_bf16.
// ---------------------------------------------------------------------------
__global__ __launch_bounds__(256)
void scores_kernel(const unsigned short* __restrict__ qbf,
                   const unsigned short* __restrict__ kbf,
                   const float* __restrict__ hw, float* __restrict__ out) {
  __shared__ unsigned short lq[64 * 72];   // padded stride 72 (anti bank-alias)
  __shared__ unsigned short lk[64 * 72];

  const int bi = blockIdx.z;
  const int i0 = blockIdx.y * 64;          // query rows
  const int j0 = blockIdx.x * 64;          // key cols
  const int tid  = threadIdx.x;
  const int wave = tid >> 5;
  const int lane = tid & 31;
  const int ti   = wave >> 1;
  const int tj0  = (wave & 1) * 2;

  const unsigned short* qb = qbf + (size_t)bi * SEQ * FAN_OUT;
  const unsigned short* kb = kbf + (size_t)bi * SEQ * FAN_OUT;

  const unsigned lq_base = lds_addr_of(lq);
  const unsigned lk_base = lds_addr_of(lk);

  v8f acc0 = {}, acc1 = {};

#pragma unroll
  for (int h = 0; h < HEADS; ++h) {
    __syncthreads();                       // prev iter LDS reads done
    // DMA 64 rows x 64 cols (head slice) of Q and K: 512 x 16B each
#pragma unroll
    for (int c = tid; c < 512; c += 256) {
      int row = c >> 3, u = c & 7;
      unsigned loff = (unsigned)(row * 72 + u * 8) * 2u;
      const unsigned short* gq = qb + ((size_t)(i0 + row) << 8) + (h << 6) + u * 8;
      const unsigned short* gk = kb + ((size_t)(j0 + row) << 8) + (h << 6) + u * 8;
      async_copy16(lq_base + loff, gq);
      async_copy16(lk_base + loff, gk);
      if (h + 1 < HEADS) {                 // warm L2/L0 for next head slice
        __builtin_prefetch(gq + HDIM, 0, 1);
        __builtin_prefetch(gk + HDIM, 0, 1);
      }
    }
    wait_async0();                         // my DMAs landed in LDS
    __syncthreads();                       // everyone's DMAs landed

    v8f s0 = {}, s1 = {};
#pragma unroll
    for (int kk = 0; kk < HDIM; kk += 32) {
      v16bf a  = ldA(lq + ti  * 16 * 72 + kk, 72, lane);
      v16bf b0 = ldB(lk + tj0 * 16 * 72 + kk, 72, lane);
      v16bf b1 = ldB(lk + (tj0 + 1) * 16 * 72 + kk, 72, lane);
      s0 = __builtin_amdgcn_wmma_f32_16x16x32_bf16(false, a, false, b0, (short)0, s0, false, false);
      s1 = __builtin_amdgcn_wmma_f32_16x16x32_bf16(false, a, false, b1, (short)0, s1, false, false);
    }

    const float wh = hw[h];
#pragma unroll
    for (int r = 0; r < 8; ++r) {
      acc0[r] += wh * fmaxf(s0[r], 0.0f);
      acc1[r] += wh * fmaxf(s1[r], 0.0f);
    }
  }

  float* op = out + (size_t)bi * SEQ * SEQ;
  const int rbase = i0 + ti * 16 + ((lane >> 4) << 3);
  const int c0 = j0 + tj0 * 16 + (lane & 15);
#pragma unroll
  for (int r = 0; r < 8; ++r) {
    op[(size_t)(rbase + r) * SEQ + c0]      = acc0[r];
    op[(size_t)(rbase + r) * SEQ + c0 + 16] = acc1[r];
  }
}

// ---------------------------------------------------------------------------
extern "C" void kernel_launch(void* const* d_in, const int* in_sizes, int n_in,
                              void* d_out, int out_size, void* d_ws, size_t ws_size,
                              hipStream_t stream) {
  const float* x  = (const float*)d_in[0];   // [2,4096,1024]
  const float* qw = (const float*)d_in[1];   // [256,1024]
  const float* qb = (const float*)d_in[2];   // [256]
  const float* kw = (const float*)d_in[3];   // [256,1024]
  const float* kb = (const float*)d_in[4];   // [256]
  const float* hw = (const float*)d_in[5];   // [4]
  float* out = (float*)d_out;                // [2,4096,4096]

  unsigned short* qbf = (unsigned short*)d_ws;                        // 4 MB
  unsigned short* kbf = qbf + (size_t)MROWS * FAN_OUT;                // 4 MB

  dim3 pgrid(MROWS / 64, FAN_OUT / 64, 1);   // 128 x 4
  proj_kernel<<<pgrid, 256, 0, stream>>>(x, qw, qb, qbf);
  proj_kernel<<<pgrid, 256, 0, stream>>>(x, kw, kb, kbf);

  dim3 sgrid(SEQ / 64, SEQ / 64, 2);         // 64 x 64 x 2
  scores_kernel<<<sgrid, 256, 0, stream>>>(qbf, kbf, hw, out);
}
